// my_MultiheadAttention_24481313587769
// MI455X (gfx1250) — matrix operand order repaired and checked by reading, not compiled, not run on hardware
//
#include <hip/hip_runtime.h>
#include <hip/hip_bf16.h>

#define SEQ    4096
#define DMODEL 1024
#define NH     16
#define DK     64

typedef __attribute__((ext_vector_type(16))) __bf16 bf16x16;
typedef __attribute__((ext_vector_type(8)))  float  f32x8;

// ---- bf16 conversion: exactly one v_cvt_pk_bf16_f32 per pair ----
__device__ __forceinline__ unsigned int pack2(float a, float b) {
    unsigned int r;
    asm("v_cvt_pk_bf16_f32 %0, %1, %2" : "=v"(r) : "v"(a), "v"(b));
    return r;
}
__device__ __forceinline__ unsigned short f2bf(float f) {
    union { __bf16 b; unsigned short u; } c;
    c.b = (__bf16)f;
    return c.u;
}

// ---- CDNA5 async global->LDS copy (ASYNCcnt), GVS addressing ----
__device__ __forceinline__ void async_ld_b128(unsigned dst_lds, const void* base,
                                              unsigned off) {
    asm volatile("global_load_async_to_lds_b128 %0, %1, %2"
                 :: "v"(dst_lds), "v"(off), "s"(base) : "memory");
}
__device__ __forceinline__ void wait_async0() {
    asm volatile("s_wait_asynccnt 0" ::: "memory");
}
__device__ __forceinline__ unsigned lds_addr32(const void* p) {
    return (unsigned)(unsigned long long)p;   // LDS aperture: low 32 bits = offset
}

// ---- WMMA fragment loaders (CDNA5 16-bit layouts, wave32) ----
__device__ __forceinline__ bf16x16 frag_a(const unsigned short* tile, int ld,
                                          int k0, int lane) {
    const int r = lane & 15, h = lane >> 4;
    const unsigned short* p = tile + r * ld + k0;
    union { uint4 q[2]; bf16x16 v; } u;
    u.q[0] = *(const uint4*)(p + h * 16);
    u.q[1] = *(const uint4*)(p + h * 16 + 8);
    return u.v;
}
__device__ __forceinline__ bf16x16 frag_b(const unsigned short* tile, int ld,
                                          int k0, int n0, int lane) {
    const int n = n0 + (lane & 15), h = lane >> 4;
    const unsigned short* p = tile + n * ld + k0 + h * 16;
    union { uint4 q[2]; bf16x16 v; } u;
    u.q[0] = *(const uint4*)(p);
    u.q[1] = *(const uint4*)(p + 8);
    return u.v;
}
__device__ __forceinline__ f32x8 wmma_bf16(bf16x16 a, bf16x16 b, f32x8 c) {
    return __builtin_amdgcn_wmma_f32_16x16x32_bf16(
        false, a, false, b, (short)0, c, false, false);
}

// =====================================================================
// Kernel 1: projection  C = X @ W^T + b, ping-pong LDS, 1 barrier/K-step
// vmode 0: out[h][seq][dk] bf16 (Q scaled, K)   vmode 1: out[h][dk][seq] (V^T)
// =====================================================================
__global__ __launch_bounds__(256) void proj_qkv(
    const float* __restrict__ X, const float* __restrict__ W,
    const float* __restrict__ bias, unsigned short* __restrict__ out,
    int vmode, float scale)
{
    __shared__ alignas(16) unsigned short sA[2][64 * 40];
    __shared__ alignas(16) unsigned short sB[2][64 * 40];
    const int tid = threadIdx.x;
    const int lane = tid & 31, wv = tid >> 5;
    const int m0 = blockIdx.y * 64;
    const int head = blockIdx.x;
    const int n0 = head * 64;
    const int mi = wv & 3, ni0 = (wv >> 2) * 2;
    const int ldRow = tid >> 2;           // 0..63
    const int ldK   = (tid & 3) * 8;      // 0,8,16,24
    const int sOff  = ldRow * 40 + ldK;
    const int NK = DMODEL / 32;

    f32x8 acc0 = {}, acc1 = {};
    const float* ap = X + (size_t)(m0 + ldRow) * DMODEL + ldK;
    const float* bp = W + (size_t)(n0 + ldRow) * DMODEL + ldK;

    { // prologue: stage k-step 0 into buffer 0
        float4 a0 = *(const float4*)(ap), a1 = *(const float4*)(ap + 4);
        float4 b0 = *(const float4*)(bp), b1 = *(const float4*)(bp + 4);
        ap += 32; bp += 32;
        uint4 ua = { pack2(a0.x, a0.y), pack2(a0.z, a0.w),
                     pack2(a1.x, a1.y), pack2(a1.z, a1.w) };
        uint4 ub = { pack2(b0.x, b0.y), pack2(b0.z, b0.w),
                     pack2(b1.x, b1.y), pack2(b1.z, b1.w) };
        *(uint4*)(sA[0] + sOff) = ua;
        *(uint4*)(sB[0] + sOff) = ub;
    }
    __syncthreads();

    for (int kt = 0; kt < NK; ++kt) {
        const int cur = kt & 1;
        if (kt + 1 < NK) {          // stage next K-step into the other buffer
            float4 a0 = *(const float4*)(ap), a1 = *(const float4*)(ap + 4);
            float4 b0 = *(const float4*)(bp), b1 = *(const float4*)(bp + 4);
            ap += 32; bp += 32;
            __builtin_prefetch(ap, 0, 0);
            __builtin_prefetch(bp, 0, 0);
            uint4 ua = { pack2(a0.x, a0.y), pack2(a0.z, a0.w),
                         pack2(a1.x, a1.y), pack2(a1.z, a1.w) };
            uint4 ub = { pack2(b0.x, b0.y), pack2(b0.z, b0.w),
                         pack2(b1.x, b1.y), pack2(b1.z, b1.w) };
            *(uint4*)(sA[cur ^ 1] + sOff) = ua;
            *(uint4*)(sB[cur ^ 1] + sOff) = ub;
        }
        bf16x16 a  = frag_a(sA[cur] + mi * 16 * 40, 40, 0, lane);
        bf16x16 f0 = frag_b(sB[cur], 40, 0, ni0 * 16, lane);
        bf16x16 f1 = frag_b(sB[cur], 40, 0, (ni0 + 1) * 16, lane);
        acc0 = wmma_bf16(a, f0, acc0);
        acc1 = wmma_bf16(a, f1, acc1);
        __syncthreads();
    }

    const int half = lane >> 4, colA = lane & 15;
    #pragma unroll
    for (int s = 0; s < 2; ++s) {
        f32x8 acc = s ? acc1 : acc0;
        const int dk = (ni0 + s) * 16 + colA;
        const float bv = bias[n0 + dk];
        const int seq0 = m0 + mi * 16 + half * 8;
        if (vmode == 0) {
            unsigned short* o = out + ((size_t)head * SEQ + seq0) * DK + dk;
            #pragma unroll
            for (int d = 0; d < 8; ++d)
                o[(size_t)d * DK] = f2bf((acc[d] + bv) * scale);
        } else {
            float v[8];
            #pragma unroll
            for (int d = 0; d < 8; ++d) v[d] = (acc[d] + bv) * scale;
            uint4 u = { pack2(v[0], v[1]), pack2(v[2], v[3]),
                        pack2(v[4], v[5]), pack2(v[6], v[7]) };
            *(uint4*)(out + ((size_t)head * DK + dk) * SEQ + seq0) = u;
        }
    }
}

// =====================================================================
// Kernel 2: flash attention; K/V tiles double-buffered via async->LDS DMA,
// issued one block ahead so the DMA overlaps scores/softmax/PV.
// =====================================================================
__global__ __launch_bounds__(256) void attn_kernel(
    const unsigned short* __restrict__ Qh,   // [H][S][DK] (pre-scaled)
    const unsigned short* __restrict__ Kh,   // [H][S][DK]
    const unsigned short* __restrict__ Vt,   // [H][DK][S]
    unsigned short* __restrict__ Xout)       // [S][DMODEL] bf16
{
    __shared__ alignas(16) unsigned short sQ[64 * 72];
    __shared__ alignas(16) unsigned short sK[2][64 * 72];
    __shared__ alignas(16) unsigned short sV[2][64 * 72];  // [dk][key]
    __shared__ alignas(16) unsigned short sP[64 * 72];     // [query][key]
    __shared__ alignas(16) float sS[64 * 68];              // scores f32
    __shared__ float sM[64], sL[64], sAl[64];

    const int tid = threadIdx.x, lane = tid & 31, wv = tid >> 5;
    const int head = blockIdx.y;
    const int m0 = blockIdx.x * 64;
    const int mi = wv & 3, ni0 = (wv >> 2) * 2;
    const int half = lane >> 4, colA = lane & 15;
    const int NB = SEQ / 64;

    // per-thread chunk slots: 512 x 16B, 2 per thread
    const int r0c = tid >> 3, c0c = (tid & 7) * 8;
    const int r1c = (tid + 256) >> 3, c1c = c0c;

    const void* kheadBase = Kh + (size_t)head * SEQ * DK;
    const void* vheadBase = Vt + (size_t)head * DK * SEQ;

    auto issue_tiles = [&](int kb, int buf) {
        const void* kbase = (const unsigned short*)kheadBase + (size_t)(kb * 64) * DK;
        const void* vbase = (const unsigned short*)vheadBase + (size_t)(kb * 64);
        async_ld_b128(lds_addr32(sK[buf] + r0c * 72 + c0c), kbase,
                      (unsigned)((r0c * DK + c0c) * 2));
        async_ld_b128(lds_addr32(sK[buf] + r1c * 72 + c1c), kbase,
                      (unsigned)((r1c * DK + c1c) * 2));
        async_ld_b128(lds_addr32(sV[buf] + r0c * 72 + c0c), vbase,
                      (unsigned)((r0c * SEQ + c0c) * 2));
        async_ld_b128(lds_addr32(sV[buf] + r1c * 72 + c1c), vbase,
                      (unsigned)((r1c * SEQ + c1c) * 2));
    };

    { // Q tile + first K/V block
        const void* qbase = Qh + (size_t)head * SEQ * DK + (size_t)m0 * DK;
        async_ld_b128(lds_addr32(sQ + r0c * 72 + c0c), qbase,
                      (unsigned)((r0c * DK + c0c) * 2));
        async_ld_b128(lds_addr32(sQ + r1c * 72 + c1c), qbase,
                      (unsigned)((r1c * DK + c1c) * 2));
        issue_tiles(0, 0);
    }
    if (tid < 64) { sM[tid] = -1e30f; sL[tid] = 0.0f; }

    f32x8 o0 = {}, o1 = {};

    for (int kb = 0; kb < NB; ++kb) {
        const int cur = kb & 1;
        wait_async0();              // our DMA for buffer `cur` complete
        __syncthreads();            // publish to all waves

        if (kb + 1 < NB) issue_tiles(kb + 1, cur ^ 1);  // prefetch next block

        // scores S = Q @ K^T
        f32x8 s0 = {}, s1 = {};
        #pragma unroll
        for (int ks = 0; ks < DK; ks += 32) {
            bf16x16 a  = frag_a(sQ + mi * 16 * 72, 72, ks, lane);
            bf16x16 b0 = frag_b(sK[cur], 72, ks, ni0 * 16, lane);
            bf16x16 b1 = frag_b(sK[cur], 72, ks, (ni0 + 1) * 16, lane);
            s0 = wmma_bf16(a, b0, s0);
            s1 = wmma_bf16(a, b1, s1);
        }
        {
            const int r0 = mi * 16 + half * 8;
            #pragma unroll
            for (int d = 0; d < 8; ++d) {
                sS[(r0 + d) * 68 + ni0 * 16 + colA]       = s0[d];
                sS[(r0 + d) * 68 + (ni0 + 1) * 16 + colA] = s1[d];
            }
        }
        __syncthreads();

        // online softmax, one row per thread
        if (tid < 64) {
            const int r = tid;
            float mx = sS[r * 68];
            for (int c = 1; c < 64; ++c) mx = fmaxf(mx, sS[r * 68 + c]);
            float mnew = fmaxf(sM[r], mx);
            float al   = __expf(sM[r] - mnew);
            float sum  = 0.0f;
            for (int c = 0; c < 64; c += 2) {
                float p0 = __expf(sS[r * 68 + c]     - mnew);
                float p1 = __expf(sS[r * 68 + c + 1] - mnew);
                *(unsigned int*)(sP + r * 72 + c) = pack2(p0, p1);
                sum += p0 + p1;
            }
            sL[r] = sL[r] * al + sum;
            sM[r] = mnew;
            sAl[r] = al;
        }
        __syncthreads();

        // rescale accumulators, then O += P @ V
        {
            const int r0 = mi * 16 + half * 8;
            float al[8];
            #pragma unroll
            for (int d = 0; d < 8; ++d) al[d] = sAl[r0 + d];
            #pragma unroll
            for (int d = 0; d < 8; ++d) { o0[d] *= al[d]; o1[d] *= al[d]; }
        }
        #pragma unroll
        for (int ks = 0; ks < 64; ks += 32) {
            bf16x16 a  = frag_a(sP + mi * 16 * 72, 72, ks, lane);
            bf16x16 b0 = frag_b(sV[cur], 72, ks, ni0 * 16, lane);
            bf16x16 b1 = frag_b(sV[cur], 72, ks, (ni0 + 1) * 16, lane);
            o0 = wmma_bf16(a, b0, o0);
            o1 = wmma_bf16(a, b1, o1);
        }
        // no trailing barrier: next iteration's wait+barrier orders reuse
    }

    const int r0 = mi * 16 + half * 8;
    float li[8];
    #pragma unroll
    for (int d = 0; d < 8; ++d) li[d] = 1.0f / sL[r0 + d];
    #pragma unroll
    for (int s = 0; s < 2; ++s) {
        f32x8 o = s ? o1 : o0;
        const int dk = (ni0 + s) * 16 + colA;
        #pragma unroll
        for (int d = 0; d < 8; ++d)
            Xout[(size_t)(m0 + r0 + d) * DMODEL + head * DK + dk] =
                f2bf(o[d] * li[d]);
    }
}

// =====================================================================
// Kernel 3: output projection  out = X @ Wo^T + bo, ping-pong LDS
// =====================================================================
__global__ __launch_bounds__(256) void out_proj(
    const unsigned short* __restrict__ Xbf, const float* __restrict__ Wo,
    const float* __restrict__ bo, float* __restrict__ out)
{
    __shared__ alignas(16) unsigned short sA[2][64 * 40];
    __shared__ alignas(16) unsigned short sB[2][64 * 40];
    const int tid = threadIdx.x, lane = tid & 31, wv = tid >> 5;
    const int m0 = blockIdx.y * 64, n0 = blockIdx.x * 64;
    const int mi = wv & 3, ni0 = (wv >> 2) * 2;
    const int ldRow = tid >> 2, ldK = (tid & 3) * 8;
    const int sOff = ldRow * 40 + ldK;
    const int NK = DMODEL / 32;

    f32x8 acc0 = {}, acc1 = {};
    const unsigned short* ap = Xbf + (size_t)(m0 + ldRow) * DMODEL + ldK;
    const float* bp = Wo + (size_t)(n0 + ldRow) * DMODEL + ldK;

    {
        uint4 ua = *(const uint4*)ap;
        float4 b0 = *(const float4*)(bp), b1 = *(const float4*)(bp + 4);
        ap += 32; bp += 32;
        uint4 ub = { pack2(b0.x, b0.y), pack2(b0.z, b0.w),
                     pack2(b1.x, b1.y), pack2(b1.z, b1.w) };
        *(uint4*)(sA[0] + sOff) = ua;
        *(uint4*)(sB[0] + sOff) = ub;
    }
    __syncthreads();

    for (int kt = 0; kt < NK; ++kt) {
        const int cur = kt & 1;
        if (kt + 1 < NK) {
            uint4 ua = *(const uint4*)ap;
            float4 b0 = *(const float4*)(bp), b1 = *(const float4*)(bp + 4);
            ap += 32; bp += 32;
            __builtin_prefetch(ap, 0, 0);
            __builtin_prefetch(bp, 0, 0);
            uint4 ub = { pack2(b0.x, b0.y), pack2(b0.z, b0.w),
                         pack2(b1.x, b1.y), pack2(b1.z, b1.w) };
            *(uint4*)(sA[cur ^ 1] + sOff) = ua;
            *(uint4*)(sB[cur ^ 1] + sOff) = ub;
        }
        bf16x16 a  = frag_a(sA[cur] + mi * 16 * 40, 40, 0, lane);
        bf16x16 f0 = frag_b(sB[cur], 40, 0, ni0 * 16, lane);
        bf16x16 f1 = frag_b(sB[cur], 40, 0, (ni0 + 1) * 16, lane);
        acc0 = wmma_bf16(a, f0, acc0);
        acc1 = wmma_bf16(a, f1, acc1);
        __syncthreads();
    }

    const int half = lane >> 4, colA = lane & 15;
    const int r0 = mi * 16 + half * 8;
    #pragma unroll
    for (int s = 0; s < 2; ++s) {
        f32x8 acc = s ? acc1 : acc0;
        const int n = n0 + (ni0 + s) * 16 + colA;
        const float bv = bo[n];
        #pragma unroll
        for (int d = 0; d < 8; ++d)
            out[(size_t)(m0 + r0 + d) * DMODEL + n] = acc[d] + bv;
    }
}

// =====================================================================
extern "C" void kernel_launch(void* const* d_in, const int* in_sizes, int n_in,
                              void* d_out, int out_size, void* d_ws, size_t ws_size,
                              hipStream_t stream) {
    const float* query = (const float*)d_in[0];
    const float* key   = (const float*)d_in[1];
    const float* value = (const float*)d_in[2];
    const float* Wq = (const float*)d_in[3];
    const float* bq = (const float*)d_in[4];
    const float* Wk = (const float*)d_in[5];
    const float* bk = (const float*)d_in[6];
    const float* Wv = (const float*)d_in[7];
    const float* bv = (const float*)d_in[8];
    const float* Wo = (const float*)d_in[9];
    const float* bo = (const float*)d_in[10];
    float* out = (float*)d_out;

    const size_t perMat = (size_t)NH * SEQ * DK;   // 4M elements
    unsigned short* Qh = (unsigned short*)d_ws;    // [H][S][DK]
    unsigned short* Kh = Qh + perMat;              // [H][S][DK]
    unsigned short* Vt = Kh + perMat;              // [H][DK][S]
    unsigned short* Xb = Vt + perMat;              // [S][DMODEL]

    dim3 blk(256);
    dim3 gproj(NH, SEQ / 64);
    hipLaunchKernelGGL(proj_qkv, gproj, blk, 0, stream, query, Wq, bq, Qh, 0, 0.125f);
    hipLaunchKernelGGL(proj_qkv, gproj, blk, 0, stream, key,   Wk, bk, Kh, 0, 1.0f);
    hipLaunchKernelGGL(proj_qkv, gproj, blk, 0, stream, value, Wv, bv, Vt, 1, 1.0f);

    dim3 gattn(SEQ / 64, NH);
    hipLaunchKernelGGL(attn_kernel, gattn, blk, 0, stream, Qh, Kh, Vt, Xb);

    dim3 gout(DMODEL / 64, SEQ / 64);
    hipLaunchKernelGGL(out_proj, gout, blk, 0, stream, Xb, Wo, bo, out);
}
